// LSTMPPOPolicy_29935922053225
// MI455X (gfx1250) — matrix operand order repaired and checked by reading, not compiled, not run on hardware
//
#include <hip/hip_runtime.h>
#include <hip/hip_bf16.h>
#include <stdint.h>

// ---------------------------------------------------------------------------
// Types for CDNA5 WMMA
// ---------------------------------------------------------------------------
typedef __bf16 bf16_t;
typedef __attribute__((ext_vector_type(16))) __bf16 v16bf;
typedef __attribute__((ext_vector_type(8)))  float  v8f;
typedef __attribute__((ext_vector_type(2)))  float  v2f;

static constexpr int Bb = 32, Tt = 512, Dd = 256, Ee = 1024, Hh = 1024, Aa = 64;
static constexpr int Mrows = Bb * Tt;          // 16384
static constexpr int G4 = 4 * Hh;              // 4096

__device__ __forceinline__ float bf2f(bf16_t x) {
    unsigned short u = __builtin_bit_cast(unsigned short, x);
    unsigned int w = ((unsigned int)u) << 16;
    return __builtin_bit_cast(float, w);
}
__device__ __forceinline__ bf16_t f2bf(float f) {
    unsigned int u = __builtin_bit_cast(unsigned int, f);
    unsigned int r = u + 0x7FFFu + ((u >> 16) & 1u);   // round-to-nearest-even
    unsigned short h = (unsigned short)(r >> 16);
    return __builtin_bit_cast(bf16_t, h);
}
__device__ __forceinline__ float sigf(float x) { return 1.0f / (1.0f + __expf(-x)); }

// ---------------------------------------------------------------------------
// CDNA5 async Global->LDS DMA (ASYNCcnt-tracked, bypasses VGPRs).
// LDS address taken from the VDST VGPR operand; per-lane 16B transfer.
// ---------------------------------------------------------------------------
__device__ __forceinline__ void async_ld_b128(uint32_t lds_off, const void* g) {
    asm volatile("global_load_async_to_lds_b128 %0, %1, off"
                 :: "v"(lds_off), "v"(g)
                 : "memory");
}
__device__ __forceinline__ void wait_async0() {
    asm volatile("s_wait_asynccnt 0" ::: "memory");
}

// ---------------------------------------------------------------------------
// WMMA fragment loaders (ISA 7.12.2 layouts, wave32)
// A: 16x32 bf16, row-major source, lda in elements.
//   lane<16 : row=lane,   K in {0..7}  (V0..3) and {16..23} (V4..7)
//   lane>=16: row=lane-16,K in {8..15} (V0..3) and {24..31} (V4..7)
// Works for global OR LDS-backed pointers (LDS -> ds_load_b128).
// ---------------------------------------------------------------------------
__device__ __forceinline__ v16bf load_a_frag(const bf16_t* A, int lda, int m0, int k0, int lane) {
    int row = lane & 15;
    int kh  = (lane >> 4) * 8;
    const bf16_t* p = A + (size_t)(m0 + row) * lda + k0 + kh;
    union { v16bf v; uint4 q[2]; } u;
    u.q[0] = *(const uint4*)(p);        // K = kh .. kh+7
    u.q[1] = *(const uint4*)(p + 16);   // K = kh+16 .. kh+23
    return u.v;
}
// B: 32x16 bf16 = column n is row n of the (N x K) weight matrix (gates = x @ W^T).
//   lane<16 : col=lane, K 0..7,16..23 ; lane>=16 : col=lane-16, K 8..15,24..31 (same as A)
__device__ __forceinline__ v16bf load_b_frag(const bf16_t* W, int ldk, int n0, int k0, int lane) {
    int col = lane & 15;
    int kh  = (lane >> 4) * 8;
    const bf16_t* p = W + (size_t)(n0 + col) * ldk + k0 + kh;
    union { v16bf v; uint4 q[2]; } u;
    u.q[0] = *(const uint4*)(p);        // K = kh .. kh+7
    u.q[1] = *(const uint4*)(p + 16);   // K = kh+16 .. kh+23
    return u.v;
}

// ---------------------------------------------------------------------------
// Elementwise helpers
// ---------------------------------------------------------------------------
__global__ void cvt_f32_bf16_k(const float* __restrict__ s, bf16_t* __restrict__ d, size_t n) {
    size_t i = (size_t)blockIdx.x * 256 + threadIdx.x;
    if (i < n) d[i] = f2bf(s[i]);
}
__global__ void init_state_k(const float* __restrict__ hxs, const float* __restrict__ cxs,
                             bf16_t* __restrict__ hbf, float* __restrict__ c, int n) {
    int i = blockIdx.x * 256 + threadIdx.x;
    if (i < n) { hbf[i] = f2bf(hxs[i]); c[i] = cxs[i]; }
}
__global__ void gate_bias_k(const float* __restrict__ bih, const float* __restrict__ bhh,
                            float* __restrict__ gb, int n) {
    int i = blockIdx.x * 256 + threadIdx.x;
    if (i < n) gb[i] = bih[i] + bhh[i];
}

// ---------------------------------------------------------------------------
// Batched GEMM:  out = act(A(MxK,bf16) @ W(NxK,bf16)^T + bias)
// One wave computes a 16(M) x 64(N) strip: A-fragment reused across 4 WMMAs.
// Ping/pong double buffering, unrolled by 2 k-chunks: each fragment buffer is
// written directly by its loads and consumed by its WMMAs -> no register
// copies, loads of one chunk overlap WMMAs of the other.
// Requires K % 64 == 0 (true here: 256, 1024).
// Block = 4 waves (4 M-tiles).  grid = (N/64, M/64).
// ---------------------------------------------------------------------------
__global__ void __launch_bounds__(128)
gemm_bf16_k(const bf16_t* __restrict__ A, const bf16_t* __restrict__ W,
            const float* __restrict__ bias, bf16_t* __restrict__ outBf,
            float* __restrict__ outF, int M, int N, int K, int act) {
    const int lane = threadIdx.x & 31;
    const int wave = threadIdx.x >> 5;
    const int m0 = (blockIdx.y * 4 + wave) * 16;
    const int n0 = blockIdx.x * 64;
    if (m0 >= M) return;

    v8f acc[4] = {};
    v16bf a0 = load_a_frag(A, K, m0, 0, lane);
    v16bf b0[4], b1[4];
    v16bf a1;
#pragma unroll
    for (int j = 0; j < 4; ++j) b0[j] = load_b_frag(W, K, n0 + 16 * j, 0, lane);

    for (int k0 = 32; k0 < K - 32; k0 += 64) {
        a1 = load_a_frag(A, K, m0, k0, lane);
#pragma unroll
        for (int j = 0; j < 4; ++j) b1[j] = load_b_frag(W, K, n0 + 16 * j, k0, lane);
#pragma unroll
        for (int j = 0; j < 4; ++j)
            acc[j] = __builtin_amdgcn_wmma_f32_16x16x32_bf16(false, a0, false, b0[j],
                                                             (short)0, acc[j], false, false);
        a0 = load_a_frag(A, K, m0, k0 + 32, lane);
#pragma unroll
        for (int j = 0; j < 4; ++j) b0[j] = load_b_frag(W, K, n0 + 16 * j, k0 + 32, lane);
#pragma unroll
        for (int j = 0; j < 4; ++j)
            acc[j] = __builtin_amdgcn_wmma_f32_16x16x32_bf16(false, a1, false, b1[j],
                                                             (short)0, acc[j], false, false);
    }
    // tail: chunks K-64 (in buf0) already pending? No: buf0 holds K-64's data
    // only when loop exited after loading it; by construction buf0 holds the
    // second-to-last chunk and the last chunk remains. Load last into buf1.
    a1 = load_a_frag(A, K, m0, K - 32, lane);
#pragma unroll
    for (int j = 0; j < 4; ++j) b1[j] = load_b_frag(W, K, n0 + 16 * j, K - 32, lane);
#pragma unroll
    for (int j = 0; j < 4; ++j)
        acc[j] = __builtin_amdgcn_wmma_f32_16x16x32_bf16(false, a0, false, b0[j],
                                                         (short)0, acc[j], false, false);
#pragma unroll
    for (int j = 0; j < 4; ++j)
        acc[j] = __builtin_amdgcn_wmma_f32_16x16x32_bf16(false, a1, false, b1[j],
                                                         (short)0, acc[j], false, false);

    const int rb  = (lane >> 4) * 8;
    const int cl  = lane & 15;
#pragma unroll
    for (int j = 0; j < 4; ++j) {
#pragma unroll
        for (int r = 0; r < 8; ++r) {
            int row = m0 + rb + r;
            int col = n0 + 16 * j + cl;
            float v = acc[j][r] + bias[col];
            if (act) v = v * sigf(v);                 // SiLU
            size_t idx = (size_t)row * N + col;
            if (outF)  outF[idx]  = v;
            if (outBf) outBf[idx] = f2bf(v);
        }
    }
}

// ---------------------------------------------------------------------------
// One LSTM time step:  gates = h_{t-1} @ W_hh^T + x_gates[t];  c,h update.
// grid = H/16 blocks, block = 2 waves (M-tiles covering B=32 rows).
// h_{t-1} (32x1024 bf16 = 64KB) is staged once per block into LDS with the
// CDNA5 async Global->LDS DMA; A-fragments then come from LDS (ds_load_b128)
// for all 32 k-chunks, while the W_hh B-fragments are ping/pong-pipelined
// from L2.  Each wave computes the SAME 16-col slice of all 4 gates, so the
// gate nonlinearity is pure per-lane VALU.  h double-buffered across the 512
// step launches (stream order = inter-step barrier).
// ---------------------------------------------------------------------------
__global__ void __launch_bounds__(64)
lstm_step_k(const bf16_t* __restrict__ h_in, const bf16_t* __restrict__ Whh,
            const float* __restrict__ xg, float* __restrict__ c,
            float* __restrict__ h_all, bf16_t* __restrict__ h_out, int t) {
    __shared__ bf16_t hsh[Bb * Hh];               // 64 KB staged h_{t-1}
    const int lane = threadIdx.x & 31;
    const int wave = threadIdx.x >> 5;
    const int m0 = wave * 16;
    const int n0 = blockIdx.x * 16;

    {   // cooperative async stage: 4096 x 16B chunks across 64 threads
        uint32_t lds_base = (uint32_t)(uintptr_t)&hsh[0];
        const char* gsrc = (const char*)h_in;
        for (int i = threadIdx.x; i < (Bb * Hh * 2) / 16; i += 64)
            async_ld_b128(lds_base + (uint32_t)i * 16u, gsrc + (size_t)i * 16);
        wait_async0();
        __syncthreads();
    }

    v8f acc[4] = {};
    v16bf a0 = load_a_frag(hsh, Hh, m0, 0, lane);
    v16bf b0[4], b1[4];
    v16bf a1;
#pragma unroll
    for (int g = 0; g < 4; ++g) b0[g] = load_b_frag(Whh, Hh, g * Hh + n0, 0, lane);

    for (int k0 = 32; k0 < Hh - 32; k0 += 64) {
        a1 = load_a_frag(hsh, Hh, m0, k0, lane);
#pragma unroll
        for (int g = 0; g < 4; ++g) b1[g] = load_b_frag(Whh, Hh, g * Hh + n0, k0, lane);
#pragma unroll
        for (int g = 0; g < 4; ++g)
            acc[g] = __builtin_amdgcn_wmma_f32_16x16x32_bf16(false, a0, false, b0[g],
                                                             (short)0, acc[g], false, false);
        a0 = load_a_frag(hsh, Hh, m0, k0 + 32, lane);
#pragma unroll
        for (int g = 0; g < 4; ++g) b0[g] = load_b_frag(Whh, Hh, g * Hh + n0, k0 + 32, lane);
#pragma unroll
        for (int g = 0; g < 4; ++g)
            acc[g] = __builtin_amdgcn_wmma_f32_16x16x32_bf16(false, a1, false, b1[g],
                                                             (short)0, acc[g], false, false);
    }
    a1 = load_a_frag(hsh, Hh, m0, Hh - 32, lane);
#pragma unroll
    for (int g = 0; g < 4; ++g) b1[g] = load_b_frag(Whh, Hh, g * Hh + n0, Hh - 32, lane);
#pragma unroll
    for (int g = 0; g < 4; ++g)
        acc[g] = __builtin_amdgcn_wmma_f32_16x16x32_bf16(false, a0, false, b0[g],
                                                         (short)0, acc[g], false, false);
#pragma unroll
    for (int g = 0; g < 4; ++g)
        acc[g] = __builtin_amdgcn_wmma_f32_16x16x32_bf16(false, a1, false, b1[g],
                                                         (short)0, acc[g], false, false);

    const int rb  = (lane >> 4) * 8;
    const int col = n0 + (lane & 15);
#pragma unroll
    for (int r = 0; r < 8; ++r) {
        int brow = m0 + rb + r;
        size_t xrow = ((size_t)brow * Tt + t) * (size_t)G4;
        float gi = acc[0][r] + xg[xrow + 0 * Hh + col];
        float gf = acc[1][r] + xg[xrow + 1 * Hh + col];
        float gg = acc[2][r] + xg[xrow + 2 * Hh + col];
        float go = acc[3][r] + xg[xrow + 3 * Hh + col];
        float cv = sigf(gf) * c[(size_t)brow * Hh + col] + sigf(gi) * tanhf(gg);
        c[(size_t)brow * Hh + col] = cv;
        float hv = sigf(go) * tanhf(cv);
        h_all[((size_t)brow * Tt + t) * Hh + col] = hv;
        h_out[(size_t)brow * Hh + col] = f2bf(hv);
    }
}

// ---------------------------------------------------------------------------
// LayerNorm per (b,t) row + per-row sum(out_ln^2) for the AR loss.
// ---------------------------------------------------------------------------
__global__ void __launch_bounds__(256)
layernorm_k(const float* __restrict__ x, const float* __restrict__ g,
            const float* __restrict__ bt, float* __restrict__ y,
            float* __restrict__ row_ar) {
    __shared__ float s1[256], s2[256];
    const int row = blockIdx.x;
    const float* xr = x + (size_t)row * Hh;
    float sum = 0.f, sq = 0.f;
    for (int j = threadIdx.x; j < Hh; j += 256) { float v = xr[j]; sum += v; sq += v * v; }
    s1[threadIdx.x] = sum; s2[threadIdx.x] = sq; __syncthreads();
    for (int s = 128; s > 0; s >>= 1) {
        if (threadIdx.x < s) { s1[threadIdx.x] += s1[threadIdx.x + s]; s2[threadIdx.x] += s2[threadIdx.x + s]; }
        __syncthreads();
    }
    float mu  = s1[0] * (1.0f / Hh);
    float var = s2[0] * (1.0f / Hh) - mu * mu;
    float rs  = rsqrtf(var + 1e-5f);
    float a2 = 0.f;
    for (int j = threadIdx.x; j < Hh; j += 256) {
        float v = (xr[j] - mu) * rs * g[j] + bt[j];
        y[(size_t)row * Hh + j] = v;
        a2 += v * v;
    }
    __syncthreads();
    s1[threadIdx.x] = a2; __syncthreads();
    for (int s = 128; s > 0; s >>= 1) {
        if (threadIdx.x < s) s1[threadIdx.x] += s1[threadIdx.x + s];
        __syncthreads();
    }
    if (threadIdx.x == 0) row_ar[row] = s1[0];
}

// Temporal-AR: per (b,t>=1) row sum of squared diffs of out_ln.
__global__ void __launch_bounds__(256)
tar_rows_k(const float* __restrict__ y, float* __restrict__ row_tar) {
    __shared__ float s[256];
    int idx = blockIdx.x;                 // [0, B*(T-1))
    int b = idx / (Tt - 1);
    int t = idx % (Tt - 1) + 1;
    const float* a = y + ((size_t)b * Tt + t) * Hh;
    const float* p = y + ((size_t)b * Tt + t - 1) * Hh;
    float acc = 0.f;
    for (int j = threadIdx.x; j < Hh; j += 256) { float d = a[j] - p[j]; acc += d * d; }
    s[threadIdx.x] = acc; __syncthreads();
    for (int st = 128; st > 0; st >>= 1) {
        if (threadIdx.x < st) s[threadIdx.x] += s[threadIdx.x + st];
        __syncthreads();
    }
    if (threadIdx.x == 0) row_tar[idx] = s[0];
}

// Deterministic final reduction of both losses (fixed order; no atomics).
__global__ void __launch_bounds__(256)
finalize_losses_k(const float* __restrict__ row_ar, const float* __restrict__ row_tar,
                  float* __restrict__ out_ar, float* __restrict__ out_tar,
                  int n_ar, int n_tar, float ar_scale, float tar_scale) {
    __shared__ float s[256];
    float a = 0.f;
    for (int i = threadIdx.x; i < n_ar; i += 256) a += row_ar[i];
    s[threadIdx.x] = a; __syncthreads();
    for (int st = 128; st > 0; st >>= 1) { if (threadIdx.x < st) s[threadIdx.x] += s[threadIdx.x + st]; __syncthreads(); }
    if (threadIdx.x == 0) *out_ar = s[0] * ar_scale;
    __syncthreads();
    float b = 0.f;
    for (int i = threadIdx.x; i < n_tar; i += 256) b += row_tar[i];
    s[threadIdx.x] = b; __syncthreads();
    for (int st = 128; st > 0; st >>= 1) { if (threadIdx.x < st) s[threadIdx.x] += s[threadIdx.x + st]; __syncthreads(); }
    if (threadIdx.x == 0) *out_tar = s[0] * tar_scale;
}

// ---------------------------------------------------------------------------
// Actor head in FULL fp32 via V_WMMA_F32_16X16X4_F32 (final-output precision).
// A 16x4 f32: lane<16 row=lane K={0,1}; lane>=16 K={2,3}.  B 4x16 mirrored.
// One wave per 16-row strip computes all N=64 columns (4 accumulators).
// ---------------------------------------------------------------------------
__global__ void __launch_bounds__(32)
logits_f32_k(const float* __restrict__ X, const float* __restrict__ Wa,
             const float* __restrict__ ba, float* __restrict__ out, int K) {
    const int lane = threadIdx.x & 31;
    const int m0 = blockIdx.x * 16;
    const int row = lane & 15;
    const int kb  = (lane >> 4) * 2;
    v8f acc[4] = {};
    for (int k0 = 0; k0 < K; k0 += 4) {
        const float* pa = X + (size_t)(m0 + row) * K + k0 + kb;
        v2f a; a[0] = pa[0]; a[1] = pa[1];
#pragma unroll
        for (int j = 0; j < 4; ++j) {
            const float* pb = Wa + (size_t)(16 * j + row) * K + k0 + kb;  // B col = Wa row
            v2f b; b[0] = pb[0]; b[1] = pb[1];
            acc[j] = __builtin_amdgcn_wmma_f32_16x16x4_f32(false, a, false, b,
                                                           (short)0, acc[j], false, false);
        }
    }
    const int rb = (lane >> 4) * 8;
#pragma unroll
    for (int j = 0; j < 4; ++j)
#pragma unroll
        for (int r = 0; r < 8; ++r) {
            int rr = m0 + rb + r, cc = 16 * j + (lane & 15);
            out[(size_t)rr * Aa + cc] = acc[j][r] + ba[cc];
        }
}

// Critic head: one wave per row, lane-strided dot + shuffle reduce.
__global__ void __launch_bounds__(128)
value_head_k(const float* __restrict__ X, const float* __restrict__ Wc,
             const float* __restrict__ bc, float* __restrict__ out, int M, int K) {
    int wave = threadIdx.x >> 5, lane = threadIdx.x & 31;
    int row = blockIdx.x * 4 + wave;
    if (row >= M) return;
    const float* xr = X + (size_t)row * K;
    float a = 0.f;
    for (int j = lane; j < K; j += 32) a += xr[j] * Wc[j];
    for (int s = 16; s > 0; s >>= 1) a += __shfl_xor(a, s, 32);
    if (lane == 0) out[row] = a + bc[0];
}

__global__ void copy_final_state_k(const float* __restrict__ h_all, const float* __restrict__ c,
                                   float* __restrict__ hn, float* __restrict__ cn) {
    int i = blockIdx.x * 256 + threadIdx.x;  // B*H
    if (i >= Bb * Hh) return;
    int b = i / Hh, j = i % Hh;
    hn[i] = h_all[((size_t)b * Tt + Tt - 1) * Hh + j];
    cn[i] = c[i];
}

// ---------------------------------------------------------------------------
// Host orchestration
// ---------------------------------------------------------------------------
extern "C" void kernel_launch(void* const* d_in, const int* in_sizes, int n_in,
                              void* d_out, int out_size, void* d_ws, size_t ws_size,
                              hipStream_t stream) {
    const float* obs  = (const float*)d_in[0];
    const float* hxs  = (const float*)d_in[1];
    const float* cxs  = (const float*)d_in[2];
    const float* W1   = (const float*)d_in[3];
    const float* b1   = (const float*)d_in[4];
    const float* W2   = (const float*)d_in[5];
    const float* b2   = (const float*)d_in[6];
    const float* Wih  = (const float*)d_in[7];
    const float* Whh  = (const float*)d_in[8];
    const float* bih  = (const float*)d_in[9];
    const float* bhh  = (const float*)d_in[10];
    const float* ln_g = (const float*)d_in[11];
    const float* ln_b = (const float*)d_in[12];
    const float* Wa   = (const float*)d_in[13];
    const float* ba   = (const float*)d_in[14];
    const float* Wc   = (const float*)d_in[15];
    const float* bc   = (const float*)d_in[16];
    float* out = (float*)d_out;

    // Workspace carve-out (256B aligned slices).
    char* base = (char*)d_ws;
    size_t off = 0;
    auto take = [&](size_t bytes) -> char* {
        char* p = base + off;
        off = (off + bytes + 255) & ~(size_t)255;
        return p;
    };
    bf16_t* obs_bf  = (bf16_t*)take((size_t)Mrows * Dd * 2);
    bf16_t* W1_bf   = (bf16_t*)take((size_t)Ee * Dd * 2);
    bf16_t* W2_bf   = (bf16_t*)take((size_t)Ee * Ee * 2);
    bf16_t* Wih_bf  = (bf16_t*)take((size_t)G4 * Ee * 2);
    bf16_t* Whh_bf  = (bf16_t*)take((size_t)G4 * Hh * 2);
    bf16_t* enc1_bf = (bf16_t*)take((size_t)Mrows * Ee * 2);
    bf16_t* enc2_bf = (bf16_t*)take((size_t)Mrows * Ee * 2);
    float*  xg      = (float*) take((size_t)Mrows * G4 * 4);     // 256 MB
    float*  gbias   = (float*) take((size_t)G4 * 4);
    bf16_t* hbuf0   = (bf16_t*)take((size_t)Bb * Hh * 2);
    bf16_t* hbuf1   = (bf16_t*)take((size_t)Bb * Hh * 2);
    float*  cbuf    = (float*) take((size_t)Bb * Hh * 4);
    float*  h_all   = (float*) take((size_t)Mrows * Hh * 4);     // 64 MB
    float*  out_ln  = (float*) take((size_t)Mrows * Hh * 4);     // 64 MB
    float*  row_ar  = (float*) take((size_t)Mrows * 4);
    float*  row_tar = (float*) take((size_t)Bb * (Tt - 1) * 4);

    // Output section offsets (return order: logits, values, h_n, c_n, ar, tar).
    float* o_logits = out;
    float* o_values = out + (size_t)Bb * Tt * Aa;
    float* o_hn     = o_values + (size_t)Bb * Tt;
    float* o_cn     = o_hn + (size_t)Bb * Hh;
    float* o_ar     = o_cn + (size_t)Bb * Hh;
    float* o_tar    = o_ar + 1;

    auto cvt = [&](const float* s, bf16_t* d, size_t n) {
        cvt_f32_bf16_k<<<(unsigned)((n + 255) / 256), 256, 0, stream>>>(s, d, n);
    };
    // 1) one-time (per launch) bf16 conversions
    cvt(obs, obs_bf, (size_t)Mrows * Dd);
    cvt(W1,  W1_bf,  (size_t)Ee * Dd);
    cvt(W2,  W2_bf,  (size_t)Ee * Ee);
    cvt(Wih, Wih_bf, (size_t)G4 * Ee);
    cvt(Whh, Whh_bf, (size_t)G4 * Hh);
    gate_bias_k<<<(G4 + 255) / 256, 256, 0, stream>>>(bih, bhh, gbias, G4);
    init_state_k<<<(Bb * Hh + 255) / 256, 256, 0, stream>>>(hxs, cxs, hbuf0, cbuf, Bb * Hh);

    // 2) encoder GEMMs + input-gate projection (bf16 WMMA, f32 accumulate)
    gemm_bf16_k<<<dim3(Ee / 64, Mrows / 64), 128, 0, stream>>>(
        obs_bf, W1_bf, b1, enc1_bf, nullptr, Mrows, Ee, Dd, 1);
    gemm_bf16_k<<<dim3(Ee / 64, Mrows / 64), 128, 0, stream>>>(
        enc1_bf, W2_bf, b2, enc2_bf, nullptr, Mrows, Ee, Ee, 1);
    gemm_bf16_k<<<dim3(G4 / 64, Mrows / 64), 128, 0, stream>>>(
        enc2_bf, Wih_bf, gbias, nullptr, xg, Mrows, G4, Ee, 0);

    // 3) sequential LSTM scan: 512 step kernels, h double-buffered
    for (int t = 0; t < Tt; ++t) {
        bf16_t* hin  = (t & 1) ? hbuf1 : hbuf0;
        bf16_t* hout = (t & 1) ? hbuf0 : hbuf1;
        lstm_step_k<<<Hh / 16, 64, 0, stream>>>(hin, Whh_bf, xg, cbuf, h_all, hout, t);
    }

    // 4) LayerNorm + losses (deterministic two-pass reduction)
    layernorm_k<<<Mrows, 256, 0, stream>>>(h_all, ln_g, ln_b, out_ln, row_ar);
    tar_rows_k<<<Bb * (Tt - 1), 256, 0, stream>>>(out_ln, row_tar);
    finalize_losses_k<<<1, 256, 0, stream>>>(
        row_ar, row_tar, o_ar, o_tar, Mrows, Bb * (Tt - 1),
        0.01f / ((float)Mrows * Hh), 0.01f / ((float)Bb * (Tt - 1) * Hh));

    // 5) heads + final state
    logits_f32_k<<<Mrows / 16, 32, 0, stream>>>(out_ln, Wa, ba, o_logits, Hh);
    value_head_k<<<Mrows / 4, 128, 0, stream>>>(out_ln, Wc, bc, o_values, Mrows, Hh);
    copy_final_state_k<<<(Bb * Hh + 255) / 256, 256, 0, stream>>>(h_all, cbuf, o_hn, o_cn);
}
